// Net_1211180777954
// MI455X (gfx1250) — matrix operand order repaired and checked by reading, not compiled, not run on hardware
//
#include <hip/hip_runtime.h>
#include <hip/hip_bf16.h>
#include <cstdint>
#include <cstddef>

// ---------------------------------------------------------------------------
// GAT (2 layers, H=4 heads) + mean pool + FC for MI455X (gfx1250, wave32).
// GEMMs via v_wmma_f32_16x16x32_f16 (f16 inputs, f32 accumulate).
// Edge softmax/scatter via f32 atomics (accumulators fit in 192MB L2).
// ---------------------------------------------------------------------------

typedef _Float16 half8  __attribute__((ext_vector_type(8)));
typedef _Float16 half16 __attribute__((ext_vector_type(16)));
typedef float    float8 __attribute__((ext_vector_type(8)));

#define DEVINL __device__ __forceinline__

// ---- WMMA fragment loaders (ISA 7.12.2, 16-bit A 16x32 / B 32x16 layouts) --
// A (row-major [M,K]): lane l -> row = l&15, hs = l>>4.
//   elements 0..7  = K[8*hs   .. 8*hs+7]   (contiguous 16B)
//   elements 8..15 = K[16+8*hs.. 16+8*hs+7](contiguous 16B)
DEVINL half16 frag_a(const _Float16* __restrict__ row, int k0, int hs) {
    half8 lo = *(const half8*)(row + k0 + 8 * hs);
    half8 hi = *(const half8*)(row + k0 + 16 + 8 * hs);
    return __builtin_shufflevector(lo, hi, 0,1,2,3,4,5,6,7,8,9,10,11,12,13,14,15);
}
// B stored transposed as [N,K] row-major: lane l -> col = l&15, hs = l>>4.
//   elements 0..7 = K[8*hs .. 8*hs+7], elements 8..15 = K[16+8*hs .. 16+8*hs+7]
DEVINL half16 frag_b(const _Float16* __restrict__ col, int k0, int hs) {
    half8 lo = *(const half8*)(col + k0 + 8 * hs);
    half8 hi = *(const half8*)(col + k0 + 16 + 8 * hs);
    return __builtin_shufflevector(lo, hi, 0,1,2,3,4,5,6,7,8,9,10,11,12,13,14,15);
}

// ---- WMMA GEMM: D[M,N] (f32) = A[M,K] (f16) * Bt[N,K]^T (f16) ------------
// One wave computes a 16x64 strip: A fragment reused across 4 WMMA issues.
__global__ void gemm_wmma_f16(const _Float16* __restrict__ A,
                              const _Float16* __restrict__ Bt,
                              float* __restrict__ D,
                              int M, int N, int K) {
    long gid  = (long)blockIdx.x * blockDim.x + threadIdx.x;
    long wave = gid >> 5;
    int  lane = threadIdx.x & 31;
    int  nGroups = N >> 6;                 // 64-wide column groups
    int  tm = (int)(wave / nGroups);
    int  tg = (int)(wave % nGroups);
    if (tm >= (M >> 4)) return;            // wave-uniform: EXEC stays all-1s

    int r  = lane & 15;
    int hs = lane >> 4;
    const _Float16* arow = A  + (size_t)(tm * 16 + r) * K;
    const _Float16* b0p  = Bt + (size_t)(tg * 64 + r) * K;
    const _Float16* b1p  = b0p + (size_t)16 * K;
    const _Float16* b2p  = b0p + (size_t)32 * K;
    const _Float16* b3p  = b0p + (size_t)48 * K;

    float8 c0 = {}, c1 = {}, c2 = {}, c3 = {};
    for (int k0 = 0; k0 < K; k0 += 32) {
        __builtin_prefetch(arow + k0 + 32, 0, 1);   // global_prefetch_b8
        __builtin_prefetch(b0p  + k0 + 32, 0, 1);
        half16 a  = frag_a(arow, k0, hs);
        half16 b0 = frag_b(b0p,  k0, hs);
        half16 b1 = frag_b(b1p,  k0, hs);
        half16 b2 = frag_b(b2p,  k0, hs);
        half16 b3 = frag_b(b3p,  k0, hs);
        c0 = __builtin_amdgcn_wmma_f32_16x16x32_f16(false, a, false, b0, (short)0, c0, false, false);
        c1 = __builtin_amdgcn_wmma_f32_16x16x32_f16(false, a, false, b1, (short)0, c1, false, false);
        c2 = __builtin_amdgcn_wmma_f32_16x16x32_f16(false, a, false, b2, (short)0, c2, false, false);
        c3 = __builtin_amdgcn_wmma_f32_16x16x32_f16(false, a, false, b3, (short)0, c3, false, false);
    }
    // C/D layout: VGPR v, lane l -> row M = v + 8*hs, col N = l&15
    int col = tg * 64 + (lane & 15);
    #pragma unroll
    for (int v = 0; v < 8; ++v) {
        size_t rowoff = (size_t)(tm * 16 + 8 * hs + v) * N;
        D[rowoff + col]      = c0[v];
        D[rowoff + col + 16] = c1[v];
        D[rowoff + col + 32] = c2[v];
        D[rowoff + col + 48] = c3[v];
    }
}

// ---- elementwise / conversion helpers -------------------------------------
__global__ void fill_f32(float* __restrict__ p, float v, long n) {
    long i = (long)blockIdx.x * blockDim.x + threadIdx.x;
    if (i < n) p[i] = v;
}
__global__ void cvt_f32_to_f16(const float* __restrict__ s, _Float16* __restrict__ d, long n) {
    long i = (long)blockIdx.x * blockDim.x + threadIdx.x;
    if (i < n) d[i] = (_Float16)s[i];
}
// W[K,N] f32 -> Wt[N,K] f16 (transposed for contiguous B fragments)
__global__ void transpose_to_f16(const float* __restrict__ W, _Float16* __restrict__ Wt,
                                 int K, int N) {
    long i = (long)blockIdx.x * blockDim.x + threadIdx.x;
    long total = (long)K * N;
    if (i >= total) return;
    int n = (int)(i / K);
    int k = (int)(i % K);
    Wt[i] = (_Float16)W[(long)k * N + n];
}

// ---- alpha_src / alpha_dst: one wave per (node, head), C-dim reduction ----
__global__ void alpha_kernel(const float* __restrict__ xw,
                             const float* __restrict__ a_src,
                             const float* __restrict__ a_dst,
                             float* __restrict__ as, float* __restrict__ ad,
                             long Nn, int H, int C) {
    long gid  = (long)blockIdx.x * blockDim.x + threadIdx.x;
    long wave = gid >> 5;
    int  lane = threadIdx.x & 31;
    long total = Nn * H;
    if (wave >= total) return;
    long n = wave / H;
    int  h = (int)(wave % H);
    const float* xr  = xw + (n * H + h) * (long)C;
    const float* asv = a_src + (long)h * C;
    const float* adv = a_dst + (long)h * C;
    float s = 0.f, d = 0.f;
    for (int c = lane; c < C; c += 32) {
        float v = xr[c];
        s += v * asv[c];
        d += v * adv[c];
    }
    #pragma unroll
    for (int off = 16; off > 0; off >>= 1) {
        s += __shfl_xor(s, off, 32);
        d += __shfl_xor(d, off, 32);
    }
    if (lane == 0) { as[wave] = s; ad[wave] = d; }
}

// ---- edge helpers ---------------------------------------------------------
DEVINL void edge_nodes(const int* __restrict__ ei, int E, int eidx, int& s, int& d) {
    if (eidx < E) { s = ei[eidx]; d = ei[E + eidx]; }
    else          { s = d = eidx - E; }          // implicit self-loops
}
DEVINL float leaky(float t) { return t > 0.f ? t : 0.2f * t; }
// sign-aware float atomic max (works across +/- with -1e30 init)
DEVINL void atomicMaxFloat(float* addr, float value) {
    if (value >= 0.f) atomicMax((int*)addr, __float_as_int(value));
    else              atomicMin((unsigned int*)addr, __float_as_uint(value));
}

__global__ void edge_max_k(const int* __restrict__ ei, int E, int Etot, int H,
                           const float* __restrict__ as, const float* __restrict__ ad,
                           float* __restrict__ m) {
    long gid = (long)blockIdx.x * blockDim.x + threadIdx.x;
    long total = (long)Etot * H;
    if (gid >= total) return;
    int eidx = (int)(gid / H), h = (int)(gid % H);
    int s, d; edge_nodes(ei, E, eidx, s, d);
    float e = leaky(as[(long)s * H + h] + ad[(long)d * H + h]);
    atomicMaxFloat(&m[(long)d * H + h], e);
}
__global__ void edge_exp_k(const int* __restrict__ ei, int E, int Etot, int H,
                           const float* __restrict__ as, const float* __restrict__ ad,
                           const float* __restrict__ m, float* __restrict__ den,
                           float* __restrict__ alpha_e) {
    long gid = (long)blockIdx.x * blockDim.x + threadIdx.x;
    long total = (long)Etot * H;
    if (gid >= total) return;
    int eidx = (int)(gid / H), h = (int)(gid % H);
    int s, d; edge_nodes(ei, E, eidx, s, d);
    float e  = leaky(as[(long)s * H + h] + ad[(long)d * H + h]);
    float ex = __expf(e - m[(long)d * H + h]);
    atomicAdd(&den[(long)d * H + h], ex);
    alpha_e[gid] = ex;                            // gid == eidx*H + h
}
__global__ void edge_norm_k(const int* __restrict__ ei, int E, int Etot, int H,
                            const float* __restrict__ den, float* __restrict__ alpha_e) {
    long gid = (long)blockIdx.x * blockDim.x + threadIdx.x;
    long total = (long)Etot * H;
    if (gid >= total) return;
    int eidx = (int)(gid / H), h = (int)(gid % H);
    int s, d; edge_nodes(ei, E, eidx, s, d);
    (void)s;
    alpha_e[gid] = alpha_e[gid] / (den[(long)d * H + h] + 1e-16f);
}
// scatter: acc[dst,h,:] += alpha * xw[src,h,:], float4-vectorized atomics
__global__ void edge_scatter_k(const int* __restrict__ ei, int E, int Etot, int H, int C,
                               const float* __restrict__ alpha_e,
                               const float* __restrict__ xw, float* __restrict__ acc) {
    long gid = (long)blockIdx.x * blockDim.x + threadIdx.x;
    int  CG  = C >> 2;
    long total = (long)Etot * H * CG;
    if (gid >= total) return;
    int  cg  = (int)(gid % CG);
    long t   = gid / CG;
    int  h   = (int)(t % H);
    int  eidx = (int)(t / H);
    int s, d; edge_nodes(ei, E, eidx, s, d);
    float a = alpha_e[(long)eidx * H + h];
    const float4 v = *(const float4*)(xw + ((long)s * H + h) * C + cg * 4);
    float* dstp = acc + ((long)d * H + h) * C + cg * 4;
    atomicAdd(dstp + 0, a * v.x);
    atomicAdd(dstp + 1, a * v.y);
    atomicAdd(dstp + 2, a * v.z);
    atomicAdd(dstp + 3, a * v.w);
}

// ---- head mean + bias; also emits f16 copy for the next GEMM --------------
__global__ void head_mean_k(const float* __restrict__ acc, const float* __restrict__ bias,
                            int H, int C, long Nn,
                            float* __restrict__ hout, _Float16* __restrict__ hhalf) {
    long gid = (long)blockIdx.x * blockDim.x + threadIdx.x;
    long total = Nn * C;
    if (gid >= total) return;
    long n = gid / C;
    int  c = (int)(gid % C);
    float s = 0.f;
    #pragma unroll
    for (int h = 0; h < 4; ++h) s += acc[(n * H + h) * (long)C + c];
    float v = s * (1.0f / H) + bias[c];
    hout[gid]  = v;
    hhalf[gid] = (_Float16)v;
}

// ---- batch mean pool + FC head --------------------------------------------
__global__ void pool_sum_k(const float* __restrict__ h, const int* __restrict__ batch,
                           float* __restrict__ pooled, float* __restrict__ cnt,
                           long Nn, int C) {
    long gid = (long)blockIdx.x * blockDim.x + threadIdx.x;
    long total = Nn * C;
    if (gid >= total) return;
    long n = gid / C;
    int  c = (int)(gid % C);
    int  b = batch[n];
    atomicAdd(&pooled[(long)b * C + c], h[gid]);
    if (c == 0) atomicAdd(&cnt[b], 1.0f);
}
__global__ void final_fc_k(const float* __restrict__ pooled, const float* __restrict__ cnt,
                           const float* __restrict__ fcW, const float* __restrict__ fcb,
                           float* __restrict__ out, int Bn, int C, int O) {
    int gid = blockIdx.x * blockDim.x + threadIdx.x;
    if (gid >= Bn * O) return;
    int b = gid / O, j = gid % O;
    float cc = fmaxf(cnt[b], 1.0f);
    float s = 0.f;
    for (int c = 0; c < C; ++c) s += pooled[(long)b * C + c] * fcW[(long)c * O + j];
    out[gid] = s / cc + fcb[j];
}

// ---------------------------------------------------------------------------
extern "C" void kernel_launch(void* const* d_in, const int* in_sizes, int n_in,
                              void* d_out, int out_size, void* d_ws, size_t ws_size,
                              hipStream_t stream) {
    const float* x      = (const float*)d_in[0];
    const int*   ei     = (const int*)d_in[1];
    const int*   batch  = (const int*)d_in[2];
    const float* W1     = (const float*)d_in[3];
    const float* a_src1 = (const float*)d_in[4];
    const float* a_dst1 = (const float*)d_in[5];
    const float* b1     = (const float*)d_in[6];
    const float* W2     = (const float*)d_in[7];
    const float* a_src2 = (const float*)d_in[8];
    const float* a_dst2 = (const float*)d_in[9];
    const float* b2     = (const float*)d_in[10];
    const float* fcW    = (const float*)d_in[11];
    const float* fcb    = (const float*)d_in[12];
    float* out = (float*)d_out;

    const long Nn  = in_sizes[2];            // 20000 nodes
    const int  E   = in_sizes[1] / 2;        // 160000 edges
    const int  Cin = (int)(in_sizes[0] / Nn);// 768
    const int  H = 4, C1 = 512, C2 = 256, O = 2;
    const int  Bn  = out_size / O;           // 128 graphs
    const int  Etot = E + (int)Nn;           // + self-loops
    const int  N1 = H * C1;                  // 2048
    const int  N2 = H * C2;                  // 1024

    // ---- carve workspace (256B aligned) ----
    char* wsp = (char*)d_ws;
    size_t off = 0;
    auto carve = [&](size_t bytes) -> void* {
        void* p = wsp + off;
        off += (bytes + 255) & ~(size_t)255;
        return p;
    };
    _Float16* xh   = (_Float16*)carve(sizeof(_Float16) * (size_t)Nn * Cin); // acts as h-f16 later
    _Float16* Wt   = (_Float16*)carve(sizeof(_Float16) * (size_t)N1 * Cin); // >= N2*C1
    float* xw      = (float*)carve(sizeof(float) * (size_t)Nn * N1);
    float* acc     = (float*)carve(sizeof(float) * (size_t)Nn * N1);
    float* hbuf    = (float*)carve(sizeof(float) * (size_t)Nn * C1);
    float* as      = (float*)carve(sizeof(float) * (size_t)Nn * H);
    float* ad      = (float*)carve(sizeof(float) * (size_t)Nn * H);
    float* mmax    = (float*)carve(sizeof(float) * (size_t)Nn * H);
    float* den     = (float*)carve(sizeof(float) * (size_t)Nn * H);
    float* alpha_e = (float*)carve(sizeof(float) * (size_t)Etot * H);
    float* pooled  = (float*)carve(sizeof(float) * (size_t)Bn * C2);
    float* cnt     = (float*)carve(sizeof(float) * (size_t)Bn);
    (void)ws_size; (void)n_in;

    const int TB = 256;
    auto nb = [](long n, int tb) { return (unsigned)((n + tb - 1) / tb); };

    // ================= Layer 1 =================
    cvt_f32_to_f16<<<nb(Nn * Cin, TB), TB, 0, stream>>>(x, xh, Nn * Cin);
    transpose_to_f16<<<nb((long)Cin * N1, TB), TB, 0, stream>>>(W1, Wt, Cin, N1);
    {
        long waves = (Nn / 16) * (long)(N1 / 64);
        gemm_wmma_f16<<<nb(waves * 32, TB), TB, 0, stream>>>(xh, Wt, xw, (int)Nn, N1, Cin);
    }
    alpha_kernel<<<nb(Nn * H * 32, TB), TB, 0, stream>>>(xw, a_src1, a_dst1, as, ad, Nn, H, C1);
    fill_f32<<<nb(Nn * H, TB), TB, 0, stream>>>(mmax, -1e30f, Nn * H);
    fill_f32<<<nb(Nn * H, TB), TB, 0, stream>>>(den, 0.f, Nn * H);
    fill_f32<<<nb(Nn * (long)N1, TB), TB, 0, stream>>>(acc, 0.f, Nn * (long)N1);
    edge_max_k    <<<nb((long)Etot * H, TB), TB, 0, stream>>>(ei, E, Etot, H, as, ad, mmax);
    edge_exp_k    <<<nb((long)Etot * H, TB), TB, 0, stream>>>(ei, E, Etot, H, as, ad, mmax, den, alpha_e);
    edge_norm_k   <<<nb((long)Etot * H, TB), TB, 0, stream>>>(ei, E, Etot, H, den, alpha_e);
    edge_scatter_k<<<nb((long)Etot * H * (C1 / 4), TB), TB, 0, stream>>>(ei, E, Etot, H, C1, alpha_e, xw, acc);
    head_mean_k<<<nb(Nn * C1, TB), TB, 0, stream>>>(acc, b1, H, C1, Nn, hbuf, xh);

    // ================= Layer 2 =================
    transpose_to_f16<<<nb((long)C1 * N2, TB), TB, 0, stream>>>(W2, Wt, C1, N2);
    {
        long waves = (Nn / 16) * (long)(N2 / 64);
        gemm_wmma_f16<<<nb(waves * 32, TB), TB, 0, stream>>>(xh, Wt, xw, (int)Nn, N2, C1);
    }
    alpha_kernel<<<nb(Nn * H * 32, TB), TB, 0, stream>>>(xw, a_src2, a_dst2, as, ad, Nn, H, C2);
    fill_f32<<<nb(Nn * H, TB), TB, 0, stream>>>(mmax, -1e30f, Nn * H);
    fill_f32<<<nb(Nn * H, TB), TB, 0, stream>>>(den, 0.f, Nn * H);
    fill_f32<<<nb(Nn * (long)N2, TB), TB, 0, stream>>>(acc, 0.f, Nn * (long)N2);
    edge_max_k    <<<nb((long)Etot * H, TB), TB, 0, stream>>>(ei, E, Etot, H, as, ad, mmax);
    edge_exp_k    <<<nb((long)Etot * H, TB), TB, 0, stream>>>(ei, E, Etot, H, as, ad, mmax, den, alpha_e);
    edge_norm_k   <<<nb((long)Etot * H, TB), TB, 0, stream>>>(ei, E, Etot, H, den, alpha_e);
    edge_scatter_k<<<nb((long)Etot * H * (C2 / 4), TB), TB, 0, stream>>>(ei, E, Etot, H, C2, alpha_e, xw, acc);
    head_mean_k<<<nb(Nn * C2, TB), TB, 0, stream>>>(acc, b2, H, C2, Nn, hbuf, xh);

    // ================= Pool + FC =================
    fill_f32<<<nb((long)Bn * C2, TB), TB, 0, stream>>>(pooled, 0.f, (long)Bn * C2);
    fill_f32<<<nb(Bn, TB), TB, 0, stream>>>(cnt, 0.f, Bn);
    pool_sum_k<<<nb(Nn * C2, TB), TB, 0, stream>>>(hbuf, batch, pooled, cnt, Nn, C2);
    final_fc_k<<<nb((long)Bn * O, TB), TB, 0, stream>>>(pooled, cnt, fcW, fcb, out, Bn, C2, O);
}
